// BlockPGA_21294447853672
// MI455X (gfx1250) — compile-verified, matrix operand order checked by reading, not run on hardware
//
#include <hip/hip_runtime.h>
#include <hip/hip_bf16.h>

// ---------------------------------------------------------------------------
// BlockPGA for MI455X (gfx1250, wave32).  All GEMMs lowered to
// v_wmma_f32_16x16x32_bf16 (bf16 in / f32 accumulate).  Attention is one
// workgroup (4 waves) per sequence, fully LDS-resident (~206KB of the 320KB
// per-WGP LDS); the scattered token gather uses the CDNA5 async data mover
// (global_load_async_to_lds_b128 + s_wait_asynccnt).  BN reductions are
// two-stage deterministic (no atomics).
// ---------------------------------------------------------------------------

#define CROP 300
#define NPIX (CROP * CROP)     // 90000
#define HALF (NPIX / 2)        // 45000
#define E_CH 64
#define C_CH 256
#define DH 32
#define BATCH 600              // HEADS * CROP
#define LSEQ 300
#define LPAD 320               // 20 tiles of 16, 10 k-steps of 32

typedef __attribute__((ext_vector_type(16))) __bf16 v16bf;
typedef __attribute__((ext_vector_type(8)))  float  v8f;

union BF16Frag { v16bf v; uint4 q[2]; unsigned short u[16]; };

__device__ __forceinline__ unsigned short f2bf(float f) {
    unsigned int u = __float_as_uint(f);
    unsigned int r = u + 0x7FFFu + ((u >> 16) & 1u);   // round-to-nearest-even
    return (unsigned short)(r >> 16);
}

__device__ __forceinline__ v8f wmma_bf16(v16bf a, v16bf b, v8f c) {
    // (neg_a, A, neg_b, B, c_mod, C, reuse_a, reuse_b)
    return __builtin_amdgcn_wmma_f32_16x16x32_bf16(false, a, false, b, (short)0, c, false, false);
}

// Load a 16x32 bf16 fragment (A-layout; also B when the B^T matrix is stored
// row-major with K contiguous).  Wave32 layout: lane<16 -> row=lane,
// k in {0..7, 16..23}; lane>=16 -> row=lane-16, k in {8..15, 24..31}.
__device__ __forceinline__ v16bf frag_load(const unsigned short* p0, int stride) {
    int lane = threadIdx.x & 31;
    const unsigned short* p = p0 + (size_t)(lane & 15) * stride + ((lane < 16) ? 0 : 8);
    BF16Frag f;
    f.q[0] = *(const uint4*)(p);
    f.q[1] = *(const uint4*)(p + 16);
    return f.v;
}

// Build a B fragment from channel-major fp32 x (stride NPIX between K rows).
__device__ __forceinline__ v16bf frag_load_x_strided(const float* __restrict__ x, int kbase, int n0) {
    int lane = threadIdx.x & 31;
    const float* col = x + n0 + (lane & 15);
    int kb = kbase + ((lane < 16) ? 0 : 8);
    BF16Frag f;
#pragma unroll
    for (int i = 0; i < 8; ++i) f.u[i]     = f2bf(col[(size_t)(kb + i) * NPIX]);
#pragma unroll
    for (int i = 0; i < 8; ++i) f.u[8 + i] = f2bf(col[(size_t)(kb + 16 + i) * NPIX]);
    return f.v;
}

// C/D fragment layout: VGPR v holds rows {v, v+8}; lanes 0..15 = cols of row v.
__device__ __forceinline__ void cstore_f32(float* dst, int stride, v8f c) {
    int lane = threadIdx.x & 31;
    int col = lane & 15, rb = (lane < 16) ? 0 : 8;
#pragma unroll
    for (int v = 0; v < 8; ++v) dst[(size_t)(rb + v) * stride + col] = c[v];
}
__device__ __forceinline__ void cstore_bf16(unsigned short* dst, int stride, v8f c, float scale) {
    int lane = threadIdx.x & 31;
    int col = lane & 15, rb = (lane < 16) ? 0 : 8;
#pragma unroll
    for (int v = 0; v < 8; ++v) dst[(rb + v) * stride + col] = f2bf(c[v] * scale);
}
__device__ __forceinline__ void cstore_bf16_T(unsigned short* dst, int stride, v8f c) {
    int lane = threadIdx.x & 31;
    int col = lane & 15, rb = (lane < 16) ? 0 : 8;
#pragma unroll
    for (int v = 0; v < 8; ++v) dst[col * stride + rb + v] = f2bf(c[v]);
}

// ---------------------------------------------------------------------------
// K0: stable argsort(1-prop) == (positions with prop==1 in order, then prop==0)
// via a single-workgroup running prefix scan.
// ---------------------------------------------------------------------------
__global__ void scan_kernel(const int* __restrict__ prop, int* __restrict__ obj, int* __restrict__ bg) {
    __shared__ int sh[1024];
    __shared__ int base;
    if (threadIdx.x == 0) base = 0;
    __syncthreads();
    for (int start = 0; start < NPIX; start += 1024) {
        int idx = start + (int)threadIdx.x;
        int p = (idx < NPIX) ? prop[idx] : 0;
        sh[threadIdx.x] = p;
        __syncthreads();
        for (int off = 1; off < 1024; off <<= 1) {
            int v = (threadIdx.x >= (unsigned)off) ? sh[threadIdx.x - off] : 0;
            __syncthreads();
            sh[threadIdx.x] += v;
            __syncthreads();
        }
        int ones_before = base + sh[threadIdx.x] - p;
        if (idx < NPIX) {
            if (p) obj[ones_before] = idx;
            else   bg[idx - ones_before] = idx;
        }
        __syncthreads();
        if (threadIdx.x == 0) base += sh[1023];
        __syncthreads();
    }
}

// K1: pix[h][r][c] = (r < 150 ? obj : bg)[rand_inds[h][r][c]]
__global__ void pix_kernel(const int* __restrict__ ri, const int* __restrict__ obj,
                           const int* __restrict__ bg, int* __restrict__ pix) {
    int i = blockIdx.x * 256 + threadIdx.x;
    if (i >= 2 * NPIX) return;
    int r = (i % NPIX) / CROP;
    int v = ri[i];
    pix[i] = (r < CROP / 2) ? obj[v] : bg[v];
}

// ---------------------------------------------------------------------------
// K2: conv1  h_raw(64 x 90000) = W1(64x256) @ x(256x90000).  8 waves (tile = block*4 + wave), each
// owns a 16-column spatial tile; K=256 in 8 WMMA steps; weights bf16 in LDS.
// ---------------------------------------------------------------------------
__launch_bounds__(256)
__global__ void conv1_kernel(const float* __restrict__ x, const float* __restrict__ w1,
                             float* __restrict__ h_raw) {
    __shared__ unsigned short W[E_CH * C_CH];             // 32 KB
    for (int e = threadIdx.x; e < E_CH * C_CH; e += 256) W[e] = f2bf(w1[e]);
    __syncthreads();
    int wave = threadIdx.x >> 5;
    int tile = blockIdx.x * 4 + wave;
    bool active = (tile * 16) < NPIX;
    int n0 = active ? tile * 16 : 0;
    v8f acc0 = {}, acc1 = {}, acc2 = {}, acc3 = {};
    for (int ks = 0; ks < 8; ++ks) {
        if (ks < 7)  // lowers to global_prefetch_b8
            __builtin_prefetch(x + (size_t)(ks * 32 + 32) * NPIX + n0, 0, 1);
        v16bf bx = frag_load_x_strided(x, ks * 32, n0);
        acc0 = wmma_bf16(frag_load(W + 0 * 16 * C_CH + ks * 32, C_CH), bx, acc0);
        acc1 = wmma_bf16(frag_load(W + 1 * 16 * C_CH + ks * 32, C_CH), bx, acc1);
        acc2 = wmma_bf16(frag_load(W + 2 * 16 * C_CH + ks * 32, C_CH), bx, acc2);
        acc3 = wmma_bf16(frag_load(W + 3 * 16 * C_CH + ks * 32, C_CH), bx, acc3);
    }
    if (active) {
        cstore_f32(h_raw + (size_t)0  * NPIX + n0, NPIX, acc0);
        cstore_f32(h_raw + (size_t)16 * NPIX + n0, NPIX, acc1);
        cstore_f32(h_raw + (size_t)32 * NPIX + n0, NPIX, acc2);
        cstore_f32(h_raw + (size_t)48 * NPIX + n0, NPIX, acc3);
    }
}

// ---------------------------------------------------------------------------
// BN stats: deterministic two-stage reduction (16 partials per channel).
// ---------------------------------------------------------------------------
__global__ void stats_partial(const float* __restrict__ src, float* __restrict__ pS,
                              float* __restrict__ pQ) {
    int c = blockIdx.x >> 4, part = blockIdx.x & 15;
    const float* p = src + (size_t)c * NPIX;
    float s = 0.f, q = 0.f;
    for (int i = part * 256 + threadIdx.x; i < NPIX; i += 16 * 256) {
        float v = p[i]; s += v; q += v * v;
    }
    for (int off = 16; off; off >>= 1) { s += __shfl_xor(s, off, 32); q += __shfl_xor(q, off, 32); }
    __shared__ float ss[8], qq[8];
    int wave = threadIdx.x >> 5, lane = threadIdx.x & 31;
    if (lane == 0) { ss[wave] = s; qq[wave] = q; }
    __syncthreads();
    if (threadIdx.x == 0) {
        float S = 0.f, Q = 0.f;
        for (int w = 0; w < 8; ++w) { S += ss[w]; Q += qq[w]; }
        pS[blockIdx.x] = S; pQ[blockIdx.x] = Q;
    }
}
__global__ void stats_final(const float* __restrict__ pS, const float* __restrict__ pQ,
                            float* __restrict__ sum, float* __restrict__ sumsq) {
    int c = threadIdx.x;
    if (c >= 64) return;
    float S = 0.f, Q = 0.f;
    for (int p = 0; p < 16; ++p) { S += pS[c * 16 + p]; Q += pQ[c * 16 + p]; }
    sum[c] = S; sumsq[c] = Q;
}

// K4: bn1 + relu; write transposed bf16 hT (token-major) used by attention
// gather AND as the conv2 B-operand.
__global__ void bn1_apply(const float* __restrict__ h_raw, const float* __restrict__ st,
                          const float* __restrict__ g, const float* __restrict__ b,
                          unsigned short* __restrict__ hT) {
    int n = blockIdx.x * 256 + threadIdx.x;
    if (n >= NPIX) return;
    const float invN = 1.0f / (float)NPIX;
#pragma unroll 4
    for (int c = 0; c < 64; ++c) {
        float mean = st[c] * invN;
        float var  = st[64 + c] * invN - mean * mean;
        float sc   = g[c] * rsqrtf(var + 1e-5f);
        float v    = h_raw[(size_t)c * NPIX + n];
        hT[(size_t)n * 64 + c] = f2bf(fmaxf((v - mean) * sc + b[c], 0.0f));
    }
}

// ---------------------------------------------------------------------------
// K5: attention.  One workgroup (4 waves) per sequence b = head*300 + r.
// LDS: WqT(2K) WkvT(4K) S(20K) Q(20K) K(20K) VT(20K) dots(80K f32) P(40K bf16)
//  = 210944 bytes (<320KB per-WGP LDS).
// Token gather goes memory->LDS through the CDNA5 async data mover (GV mode:
// per-lane global address in VGPR pair, per-lane LDS byte address in VGPR),
// completion via ASYNCcnt.
// ---------------------------------------------------------------------------
#define ATTN_SMEM 210944
__launch_bounds__(128)
__global__ void attn_kernel(const unsigned short* __restrict__ hT, const int* __restrict__ pix,
                            const float* __restrict__ wq, const float* __restrict__ wkv,
                            float* __restrict__ o_buf) {
    extern __shared__ char smem[];
    unsigned short* WqT  = (unsigned short*)smem;            // [32][32]  B^T of Wq
    unsigned short* WkvT = WqT + 32 * 32;                    // [64][32]  B^T of Wkv
    unsigned short* S    = WkvT + 64 * 32;                   // [LPAD][32]
    unsigned short* Qm   = S + LPAD * 32;                    // [LPAD][32]
    unsigned short* Km   = Qm + LPAD * 32;                   // [LPAD][32]
    unsigned short* VT   = Km + LPAD * 32;                   // [32][LPAD]
    float*          dots = (float*)(VT + 32 * LPAD);         // [4][16][LPAD]
    unsigned short* pbf  = (unsigned short*)(dots + 4 * 16 * LPAD); // [4][16][LPAD]

    int b = blockIdx.x;
    int head = b / CROP;
    int tid = threadIdx.x, wave = tid >> 5, lane = tid & 31;

    // zero the padded tail rows (tokens 300..319) with plain LDS stores
    for (int e = tid; e < (LPAD - LSEQ) * 32; e += 128) S[LSEQ * 32 + e] = 0;

    // async token gather: 4 lanes x 16B per token, hT[pix[j]] -> LDS S[j].
    // global_load_async_to_lds_b128: VDST = per-lane LDS byte address,
    // VADDR = per-lane 64-bit global address (GV mode), tracked by ASYNCcnt.
    const int* pixb = pix + (size_t)b * CROP;
    for (int t0 = 0; t0 < LSEQ; t0 += 32) {
        int j = t0 + (tid >> 2), sub = tid & 3;
        if (j < LSEQ) {
            int n = pixb[j];
            unsigned int lds_addr =
                (unsigned int)(size_t)(S + j * 32 + sub * 8);   // flat low 32 = LDS byte addr
            unsigned long long gaddr =
                (unsigned long long)(size_t)(hT + (size_t)n * 64 + head * 32 + sub * 8);
            asm volatile("global_load_async_to_lds_b128 %0, %1, off"
                         :: "v"(lds_addr), "v"(gaddr)
                         : "memory");
        }
    }

    // overlap: load transposed weights while the async DMA is in flight
    for (int e = tid; e < 32 * 32; e += 128) { int n = e >> 5, k = e & 31; WqT[e]  = f2bf(wq[k * 32 + n]); }
    for (int e = tid; e < 64 * 32; e += 128) { int n = e >> 5, k = e & 31; WkvT[e] = f2bf(wkv[k * 64 + n]); }

    asm volatile("s_wait_asynccnt 0" ::: "memory");
    __syncthreads();

    // --- QKV projections: 6 WMMAs per 16-token tile (K=32 == DH) ---
    const float qscale = 0.17677669529663687f;  // DH^-0.5 folded into Q
    v16bf bq0 = frag_load(WqT, 32),            bq1 = frag_load(WqT + 16 * 32, 32);
    v16bf bk0 = frag_load(WkvT, 32),           bk1 = frag_load(WkvT + 16 * 32, 32);
    v16bf bv0 = frag_load(WkvT + 32 * 32, 32), bv1 = frag_load(WkvT + 48 * 32, 32);
    for (int t = wave; t < LPAD / 16; t += 4) {
        int t0 = t * 16;
        v16bf a = frag_load(S + t0 * 32, 32);
        v8f z = {};
        v8f q0 = wmma_bf16(a, bq0, z), q1 = wmma_bf16(a, bq1, z);
        v8f k0 = wmma_bf16(a, bk0, z), k1 = wmma_bf16(a, bk1, z);
        v8f v0 = wmma_bf16(a, bv0, z), v1 = wmma_bf16(a, bv1, z);
        cstore_bf16(Qm + t0 * 32,      32, q0, qscale);
        cstore_bf16(Qm + t0 * 32 + 16, 32, q1, qscale);
        cstore_bf16(Km + t0 * 32,      32, k0, 1.0f);
        cstore_bf16(Km + t0 * 32 + 16, 32, k1, 1.0f);
        cstore_bf16_T(VT + 0  * LPAD + t0, LPAD, v0);   // V stored transposed
        cstore_bf16_T(VT + 16 * LPAD + t0, LPAD, v1);
    }
    __syncthreads();

    // --- dots -> softmax -> O, one 16-row panel at a time per wave ---
    float*          dw = dots + wave * 16 * LPAD;
    unsigned short* pw = pbf  + wave * 16 * LPAD;
    for (int t = wave; t < LPAD / 16; t += 4) {
        int t0 = t * 16;
        v16bf aq = frag_load(Qm + t0 * 32, 32);
        for (int j = 0; j < LPAD / 16; ++j) {            // 20 WMMAs: Q(16x32)·K^T(32x16)
            v8f z = {};
            v8f d = wmma_bf16(aq, frag_load(Km + j * 16 * 32, 32), z);
            cstore_f32(dw + j * 16, LPAD, d);
        }
        for (int r = 0; r < 16; ++r) {                   // wave32 shfl softmax
            float* row = dw + r * LPAD;
            float m = -3.4e38f;
            for (int c = lane; c < LSEQ; c += 32) m = fmaxf(m, row[c]);
            for (int off = 16; off; off >>= 1) m = fmaxf(m, __shfl_xor(m, off, 32));
            float s = 0.f;
            for (int c = lane; c < LPAD; c += 32) {
                float e = (c < LSEQ) ? __expf(row[c] - m) : 0.f;
                row[c] = e; s += e;
            }
            for (int off = 16; off; off >>= 1) s += __shfl_xor(s, off, 32);
            float inv = 1.0f / s;
            unsigned short* prow = pw + r * LPAD;
            for (int c = lane; c < LPAD; c += 32) prow[c] = f2bf(row[c] * inv);
        }
#pragma unroll
        for (int d = 0; d < 2; ++d) {                    // O = P(16x320)·V(320x16): 10 k-steps
            v8f acc = {};
            for (int ks = 0; ks < LPAD / 32; ++ks)
                acc = wmma_bf16(frag_load(pw + ks * 32, LPAD),
                                frag_load(VT + d * 16 * LPAD + ks * 32, LPAD), acc);
            int col = lane & 15, rb = (lane < 16) ? 0 : 8;
#pragma unroll
            for (int v = 0; v < 8; ++v) {
                int row = t0 + rb + v;
                if (row < LSEQ)
                    o_buf[((size_t)b * LSEQ + row) * DH + d * 16 + col] = acc[v];
            }
        }
    }
}

// K6: the reference's reshape-scramble + permutation scatter:
// newT[pix[b,c]][head*32+d] = o[b].flat[d*300+c]   (token-major bf16 output)
__global__ void scatter_kernel(const float* __restrict__ o_buf, const int* __restrict__ pix,
                               unsigned short* __restrict__ newT) {
    int id = blockIdx.x * 256 + threadIdx.x;
    if (id >= BATCH * CROP) return;
    int b = id / CROP, cc = id % CROP;
    int head = b / CROP;
    int n = pix[(size_t)b * CROP + cc];
    const float* src = o_buf + (size_t)b * (DH * CROP) + cc;
    unsigned short* dst = newT + (size_t)n * 64 + head * 32;
#pragma unroll
    for (int d = 0; d < DH; ++d) dst[d] = f2bf(src[d * CROP]);
}

// ---------------------------------------------------------------------------
// K8: fused  t = relu(Wout^T @ new + b_out);  out_raw = W2a@t + W2b@h.
// 24 WMMAs per 16 spatial columns; mid tensor t lives only in LDS.
// ---------------------------------------------------------------------------
__launch_bounds__(256)
__global__ void fused_out_kernel(const unsigned short* __restrict__ newT,
                                 const unsigned short* __restrict__ hT,
                                 const float* __restrict__ w_out, const float* __restrict__ b_out,
                                 const float* __restrict__ w_conv2, float* __restrict__ out_raw) {
    __shared__ unsigned short WoT[64 * 64];   // A = w_out^T
    __shared__ unsigned short W2a[64 * 64];   // A = w_conv2[:, :64]
    __shared__ unsigned short W2b[64 * 64];   // A = w_conv2[:, 64:]
    __shared__ float bo[64];
    __shared__ unsigned short tT[4][16 * 64]; // per-wave t transposed (spatial-major)
    for (int e = threadIdx.x; e < 64 * 64; e += 256) {
        int r = e >> 6, c = e & 63;
        WoT[e] = f2bf(w_out[c * 64 + r]);
        W2a[e] = f2bf(w_conv2[r * 128 + c]);
        W2b[e] = f2bf(w_conv2[r * 128 + 64 + c]);
    }
    if (threadIdx.x < 64) bo[threadIdx.x] = b_out[threadIdx.x];
    __syncthreads();

    int wave = threadIdx.x >> 5, lane = threadIdx.x & 31;
    int tile = blockIdx.x * 4 + wave;
    bool active = tile < (NPIX / 16);
    int n0 = active ? tile * 16 : 0;

    v16bf bn0 = frag_load(newT + (size_t)n0 * 64, 64);
    v16bf bn1 = frag_load(newT + (size_t)n0 * 64 + 32, 64);
    unsigned short* tw = tT[wave];
#pragma unroll
    for (int m = 0; m < 4; ++m) {
        v8f z = {};
        v8f c = wmma_bf16(frag_load(WoT + m * 16 * 64, 64),      bn0, z);
        c     = wmma_bf16(frag_load(WoT + m * 16 * 64 + 32, 64), bn1, c);
        int col = lane & 15, rb = (lane < 16) ? 0 : 8;
#pragma unroll
        for (int v = 0; v < 8; ++v) {
            int row = m * 16 + rb + v;
            tw[col * 64 + row] = f2bf(fmaxf(c[v] + bo[row], 0.0f));  // bias + relu
        }
    }
    // same-wave LDS RAW only (DS ops are in-order within a wave) -> no barrier
    v16bf bt0 = frag_load(tw, 64),                               bt1 = frag_load(tw + 32, 64);
    v16bf bh0 = frag_load(hT + (size_t)n0 * 64, 64),             bh1 = frag_load(hT + (size_t)n0 * 64 + 32, 64);
#pragma unroll
    for (int m = 0; m < 4; ++m) {
        v8f z = {};
        v8f c = wmma_bf16(frag_load(W2a + m * 16 * 64, 64),      bt0, z);
        c     = wmma_bf16(frag_load(W2a + m * 16 * 64 + 32, 64), bt1, c);
        c     = wmma_bf16(frag_load(W2b + m * 16 * 64, 64),      bh0, c);
        c     = wmma_bf16(frag_load(W2b + m * 16 * 64 + 32, 64), bh1, c);
        if (active) cstore_f32(out_raw + (size_t)(m * 16) * NPIX + n0, NPIX, c);
    }
}

// K10: bn2 + relu -> final output (1,64,300,300) fp32
__global__ void bn2_apply(const float* __restrict__ raw, const float* __restrict__ st,
                          const float* __restrict__ g, const float* __restrict__ b,
                          float* __restrict__ out) {
    size_t i = (size_t)blockIdx.x * 256 + threadIdx.x;
    if (i >= (size_t)E_CH * NPIX) return;
    int c = (int)(i / NPIX);
    const float invN = 1.0f / (float)NPIX;
    float mean = st[c] * invN;
    float var  = st[64 + c] * invN - mean * mean;
    float sc   = g[c] * rsqrtf(var + 1e-5f);
    out[i] = fmaxf((raw[i] - mean) * sc + b[c], 0.0f);
}

// ---------------------------------------------------------------------------
extern "C" void kernel_launch(void* const* d_in, const int* in_sizes, int n_in,
                              void* d_out, int out_size, void* d_ws, size_t ws_size,
                              hipStream_t stream) {
    const float* x        = (const float*)d_in[0];
    const int*   prop     = (const int*)d_in[1];
    const int*   randinds = (const int*)d_in[2];
    const float* w_conv1  = (const float*)d_in[3];
    const float* bn1_g    = (const float*)d_in[4];
    const float* bn1_b    = (const float*)d_in[5];
    const float* wq       = (const float*)d_in[6];
    const float* wkv      = (const float*)d_in[7];
    const float* w_out    = (const float*)d_in[8];
    const float* b_out    = (const float*)d_in[9];
    const float* w_conv2  = (const float*)d_in[10];
    const float* bn2_g    = (const float*)d_in[11];
    const float* bn2_b    = (const float*)d_in[12];
    float* out = (float*)d_out;

    // workspace carve-out (256B aligned)
    char* ws = (char*)d_ws;
    size_t off = 0;
    auto carve = [&](size_t bytes) { char* p = ws + off; off = (off + bytes + 255) & ~(size_t)255; return p; };
    float* stats  = (float*)carve(256 * sizeof(float));        // sums1,sq1,sums2,sq2
    float* pS1    = (float*)carve(1024 * sizeof(float));
    float* pQ1    = (float*)carve(1024 * sizeof(float));
    float* pS2    = (float*)carve(1024 * sizeof(float));
    float* pQ2    = (float*)carve(1024 * sizeof(float));
    int*   obj    = (int*)carve(HALF * sizeof(int));
    int*   bg     = (int*)carve(HALF * sizeof(int));
    int*   pix    = (int*)carve(2 * (size_t)NPIX * sizeof(int));
    float* h_raw  = (float*)carve((size_t)E_CH * NPIX * sizeof(float));
    unsigned short* hT   = (unsigned short*)carve((size_t)NPIX * 64 * sizeof(unsigned short));
    float* o_buf  = (float*)carve((size_t)BATCH * LSEQ * DH * sizeof(float));
    unsigned short* newT = (unsigned short*)carve((size_t)NPIX * 64 * sizeof(unsigned short));
    float* o_raw  = (float*)carve((size_t)E_CH * NPIX * sizeof(float));

    scan_kernel<<<1, 1024, 0, stream>>>(prop, obj, bg);
    pix_kernel<<<(2 * NPIX + 255) / 256, 256, 0, stream>>>(randinds, obj, bg, pix);

    conv1_kernel<<<(NPIX / 16 + 3) / 4, 256, 0, stream>>>(x, w_conv1, h_raw);
    stats_partial<<<1024, 256, 0, stream>>>(h_raw, pS1, pQ1);
    stats_final<<<1, 64, 0, stream>>>(pS1, pQ1, stats, stats + 64);
    bn1_apply<<<(NPIX + 255) / 256, 256, 0, stream>>>(h_raw, stats, bn1_g, bn1_b, hT);

    hipFuncSetAttribute((const void*)attn_kernel,
                        hipFuncAttributeMaxDynamicSharedMemorySize, ATTN_SMEM);
    attn_kernel<<<BATCH, 128, ATTN_SMEM, stream>>>(hT, pix, wq, wkv, o_buf);

    scatter_kernel<<<(BATCH * CROP + 255) / 256, 256, 0, stream>>>(o_buf, pix, newT);

    fused_out_kernel<<<(NPIX / 16 + 3) / 4, 256, 0, stream>>>(newT, hT, w_out, b_out, w_conv2, o_raw);
    stats_partial<<<1024, 256, 0, stream>>>(o_raw, pS2, pQ2);
    stats_final<<<1, 64, 0, stream>>>(pS2, pQ2, stats + 128, stats + 192);
    bn2_apply<<<((size_t)E_CH * NPIX + 255) / 256, 256, 0, stream>>>(o_raw, stats + 128, bn2_g, bn2_b, out);
}
